// KeOpsMultiHeadAttention_79259326480575
// MI455X (gfx1250) — matrix operand order repaired and checked by reading, not compiled, not run on hardware
//
#include <hip/hip_runtime.h>
#include <hip/hip_bf16.h>

typedef _Float16 half_t;
typedef __attribute__((ext_vector_type(16))) _Float16 v16h;
typedef __attribute__((ext_vector_type(8)))  _Float16 v8h;
typedef __attribute__((ext_vector_type(8)))  float    v8f;

#define DIM 256
#define NH  8
#define HD  32
#define B_  4
#define N_  2048
#define ROWS (B_ * N_)          // 8192

// ---- WMMA wrapper: D = A(16x32 f16) * B(32x16 f16) + C(16x16 f32) ----
__device__ __forceinline__ v8f wmma16(v16h a, v16h b, v8f c) {
    return __builtin_amdgcn_wmma_f32_16x16x32_f16(
        /*neg_a=*/false, a, /*neg_b=*/false, b,
        /*c_mod=*/(short)0, c, /*reuse_a=*/false, /*reuse_b=*/false);
}

// A-operand pattern: lane (l&15) = row M; 16 contiguous K halves starting at hf*16
// caller passes p0 = base + row*rs + kbase + hf*16  (16B aligned)
__device__ __forceinline__ v16h load_tile_A(const half_t* p0) {
    union { v16h v; v8h h[2]; } u;
    u.h[0] = *(const v8h*)(p0);
    u.h[1] = *(const v8h*)(p0 + 8);
    return u.v;
}
// B-operand pattern: lane (l&15) = col N; 16 contiguous K halves starting at hf*16
// caller passes p0 = base + col*rs + kbase + hf*16  (16B aligned)
__device__ __forceinline__ v16h load_tile_B(const half_t* p0) {
    union { v16h v; v8h h[2]; } u;
    u.h[0] = *(const v8h*)(p0);
    u.h[1] = *(const v8h*)(p0 + 8);
    return u.v;
}

// ---------------- Kernel 1: f32 -> f16 conversions + weight transposes ----------------
// xh[8192][256] = x;  WqkvT[768][256] = Wqkv^T;  WoutT[256][256] = Wout^T
__global__ void cvt_kernel(const float* __restrict__ x,
                           const float* __restrict__ Wqkv,
                           const float* __restrict__ Wout,
                           half_t* __restrict__ xh,
                           half_t* __restrict__ WqkvT,
                           half_t* __restrict__ WoutT) {
    int i = blockIdx.x * 256 + threadIdx.x;
    const int NX = ROWS * DIM;            // 2,097,152
    const int NW = DIM * 3 * DIM;         //   196,608
    if (i < NX) {
        xh[i] = (half_t)x[i];
    } else if (i < NX + NW) {
        int t = i - NX;
        int n = t >> 8, k = t & 255;      // WqkvT[n][k] = Wqkv[k][n]
        WqkvT[t] = (half_t)Wqkv[k * (3 * DIM) + n];
    } else if (i < NX + NW + DIM * DIM) {
        int t = i - NX - NW;
        int n = t >> 8, k = t & 255;      // WoutT[n][k] = Wout[k][n]
        WoutT[t] = (half_t)Wout[k * DIM + n];
    }
}

// ---------------- Kernel 2: QKV GEMM  [8192,256] x [256,768] ----------------
// one wave = one 16x16 output tile; scatter into q (scaled), k, vT layouts
__global__ void __launch_bounds__(128)
qkv_gemm_kernel(const half_t* __restrict__ xh, const half_t* __restrict__ WT,
                half_t* __restrict__ qb, half_t* __restrict__ kb,
                half_t* __restrict__ vT) {
    const int NT = 3 * DIM / 16;                       // 48
    int tile = blockIdx.x * 4 + (threadIdx.x >> 5);
    if (tile >= (ROWS / 16) * NT) return;
    int mt = tile / NT, nt = tile % NT;
    int lane = threadIdx.x & 31, hf = lane >> 4, idx = lane & 15;

    const half_t* arow = xh + (size_t)(mt * 16 + idx) * DIM + hf * 16;
    const half_t* brow = WT + (size_t)(nt * 16 + idx) * DIM + hf * 16;

    v8f acc = {};
#pragma unroll
    for (int k0 = 0; k0 < DIM; k0 += 32)
        acc = wmma16(load_tile_A(arow + k0), load_tile_B(brow + k0), acc);

    int col = nt * 16 + idx;
    int sec = col >> 8;                                // 0=q 1=k 2=v (uniform per wave)
    int c2 = col & 255, h = c2 >> 5, j = c2 & 31;
    const float scale = 0.17677669529663687f;          // 32^-0.5
#pragma unroll
    for (int r = 0; r < 8; ++r) {
        int row = mt * 16 + r + hf * 8;
        int b = row >> 11, n = row & 2047;
        size_t bh = (size_t)b * NH + h;
        float v = acc[r];
        if (sec == 0)      qb[(bh * N_ + n) * HD + j] = (half_t)(v * scale);
        else if (sec == 1) kb[(bh * N_ + n) * HD + j] = (half_t)v;
        else               vT[(bh * HD + j) * N_ + n] = (half_t)v;
    }
}

// ---------------- Kernel 3: streaming attention ----------------
// one wave = 16 query rows of one (b,h); loop key blocks of 32
__global__ void __launch_bounds__(128)
attn_kernel(const half_t* __restrict__ qb, const half_t* __restrict__ kb,
            const half_t* __restrict__ vT, half_t* __restrict__ oattn) {
    __shared__ half_t Es[4][16 * 40];                  // padded 16x32 exp tile / wave
    int wave = threadIdx.x >> 5;
    int lane = threadIdx.x & 31, hf = lane >> 4, idx = lane & 15;
    int gid = blockIdx.x * 4 + wave;                   // 0..4095
    int qt = gid & 127, bh = gid >> 7;                 // 128 q-tiles per (b,h)
    int b = bh >> 3, h = bh & 7;

    const half_t* q  = qb + (size_t)bh * N_ * HD;
    const half_t* k  = kb + (size_t)bh * N_ * HD;
    const half_t* vt = vT + (size_t)bh * HD * N_;
    half_t* es = Es[wave];

    v16h aq = load_tile_A(q + (size_t)(qt * 16 + idx) * HD + hf * 16);

    v8f num0 = {}, num1 = {};
    float dsum[8] = {0.f, 0.f, 0.f, 0.f, 0.f, 0.f, 0.f, 0.f};
    const v8f zero = {};

    for (int kb0 = 0; kb0 < N_; kb0 += 32) {
        // S = q @ k^T for 32 keys (two 16-key WMMAs)
        const half_t* kp = k + (size_t)(kb0 + idx) * HD + hf * 16;
        v8f s0 = wmma16(aq, load_tile_B(kp),            zero);
        v8f s1 = wmma16(aq, load_tile_B(kp + 16 * HD),  zero);

        // unnormalized exp (matches reference), accumulate denominator,
        // restage 16x32 E tile through LDS (D-layout -> A-layout transpose)
#pragma unroll
        for (int r = 0; r < 8; ++r) {
            float e0 = fminf(__expf(s0[r]), 65000.f);
            float e1 = fminf(__expf(s1[r]), 65000.f);
            dsum[r] += e0 + e1;
            es[(r + hf * 8) * 40 + idx]      = (half_t)e0;
            es[(r + hf * 8) * 40 + 16 + idx] = (half_t)e1;
        }
        asm volatile("s_wait_dscnt 0" ::: "memory");   // RAW: LDS stores -> b128 loads
        v16h ae = load_tile_A(es + idx * 40 + hf * 16);

        // numer += E @ V  (hd=32 -> two N-split WMMAs, K=32 keys)
        const half_t* vp = vt + (size_t)idx * N_ + kb0 + hf * 16;
        num0 = wmma16(ae, load_tile_B(vp),            num0);
        num1 = wmma16(ae, load_tile_B(vp + 16 * N_),  num1);
    }

    // reduce denominator over the 16-lane N groups, normalize, store f16
#pragma unroll
    for (int r = 0; r < 8; ++r) {
        float d = dsum[r];
        d += __shfl_xor(d, 1, 32);
        d += __shfl_xor(d, 2, 32);
        d += __shfl_xor(d, 4, 32);
        d += __shfl_xor(d, 8, 32);
        float inv = 1.0f / (d + 1e-6f);
        int n = qt * 16 + r + hf * 8;
        size_t row = (size_t)(b * N_ + n) * DIM + h * HD;
        oattn[row + idx]      = (half_t)(num0[r] * inv);
        oattn[row + 16 + idx] = (half_t)(num1[r] * inv);
    }
}

// ---------------- Kernel 4: out projection [8192,256] x [256,256] + bias ----------------
__global__ void __launch_bounds__(128)
oproj_kernel(const half_t* __restrict__ oattn, const half_t* __restrict__ WoT,
             const float* __restrict__ bout, float* __restrict__ out) {
    int tile = blockIdx.x * 4 + (threadIdx.x >> 5);
    if (tile >= (ROWS / 16) * (DIM / 16)) return;
    int mt = tile >> 4, nt = tile & 15;
    int lane = threadIdx.x & 31, hf = lane >> 4, idx = lane & 15;

    const half_t* arow = oattn + (size_t)(mt * 16 + idx) * DIM + hf * 16;
    const half_t* brow = WoT   + (size_t)(nt * 16 + idx) * DIM + hf * 16;

    v8f acc = {};
#pragma unroll
    for (int k0 = 0; k0 < DIM; k0 += 32)
        acc = wmma16(load_tile_A(arow + k0), load_tile_B(brow + k0), acc);

    int col = nt * 16 + idx;
    float bias = bout[col];
#pragma unroll
    for (int r = 0; r < 8; ++r)
        out[(size_t)(mt * 16 + r + hf * 8) * DIM + col] = acc[r] + bias;
}

extern "C" void kernel_launch(void* const* d_in, const int* in_sizes, int n_in,
                              void* d_out, int out_size, void* d_ws, size_t ws_size,
                              hipStream_t stream) {
    (void)in_sizes; (void)n_in; (void)out_size; (void)ws_size;
    const float* x    = (const float*)d_in[0];   // [4,2048,256]
    const float* Wqkv = (const float*)d_in[1];   // [256,768]
    const float* Wout = (const float*)d_in[2];   // [256,256]
    const float* bout = (const float*)d_in[3];   // [256]
    float* out = (float*)d_out;                  // [4,2048,256]

    // workspace layout (halves)
    half_t* ws    = (half_t*)d_ws;
    half_t* xh    = ws;                          // 2,097,152
    half_t* WqkvT = xh + (size_t)ROWS * DIM;     //   196,608
    half_t* WoutT = WqkvT + DIM * 3 * DIM;       //    65,536
    half_t* qb    = WoutT + DIM * DIM;           // 2,097,152
    half_t* kb    = qb + (size_t)ROWS * DIM / 8 * 8; // 2,097,152 (b*h*n*hd)
    half_t* vT    = kb + (size_t)B_ * NH * N_ * HD;
    half_t* oattn = xh;                          // reuse xh after QKV GEMM

    // fix kb offset precisely (qb size = b*h*n*hd = 2,097,152)
    kb = qb + (size_t)B_ * NH * N_ * HD;
    vT = kb + (size_t)B_ * NH * N_ * HD;

    // 1) conversions / transposes: 2,359,296 elements
    cvt_kernel<<<9216, 256, 0, stream>>>(x, Wqkv, Wout, xh, WqkvT, WoutT);
    // 2) QKV GEMM: 512*48 tiles / 4 waves
    qkv_gemm_kernel<<<6144, 128, 0, stream>>>(xh, WqkvT, qb, kb, vT);
    // 3) attention: 32 (b,h) * 128 q-tiles / 4 waves
    attn_kernel<<<1024, 128, 0, stream>>>(qb, kb, vT, oattn);
    // 4) output projection: 512*16 tiles / 4 waves
    oproj_kernel<<<2048, 128, 0, stream>>>(oattn, WoutT, bout, out);
}
